// lstm_13451837571152
// MI455X (gfx1250) — matrix operand order repaired and checked by reading, not compiled, not run on hardware
//
#include <hip/hip_runtime.h>
#include <hip/hip_bf16.h>
#include <stdint.h>

// ---------------------------------------------------------------------------
// Fused 2-layer LSTM + FC for MI455X (gfx1250), wave32, WMMA f16->f32.
// B=32768, T=28, D=28 (padded to 32), H=128, gates G=4H=512, C=10.
// One 512-thread workgroup = 16 waves handles 64 batch rows for all 28 steps
// of BOTH layers (per-timestep fusion => h1 sequence never touches HBM).
// Each wave owns one 16-wide gate/h column slice and TWO 16-row batch tiles,
// so every B fragment is reused by 2 WMMAs. Gate chains seed from the WMMA
// inline-zero C operand; biases are folded into the elementwise phase
// (frees 32 VGPRs of bias splats -> no scratch spills).
// ---------------------------------------------------------------------------

typedef __attribute__((ext_vector_type(16))) _Float16 v16h;
typedef __attribute__((ext_vector_type(8)))  _Float16 v8h;
typedef __attribute__((ext_vector_type(8)))  float    v8f;
typedef __attribute__((ext_vector_type(4)))  float    v4f;

#define T_STEPS 28
#define DIMH    128

// LDS layout (bytes)
#define LDS_WHH0   0u          // [512][128] f16 = 128 KB
#define LDS_WHH1   131072u     // [512][128] f16 = 128 KB
#define LDS_H0     262144u     // [64][128]  f16 =  16 KB (layer-0 h state / layer-1 input)
#define LDS_H2     278528u     // [64][128]  f16 =  16 KB (layer-1 h state)
#define LDS_TOTAL  294912u     // 288 KB  (<= 320 KB per WGP)

// workspace (d_ws) layout (bytes) -- f16 weights + fused biases
#define WS_WIH0  0u            // [512][32]  f16 (K padded 28->32)
#define WS_WHH0  32768u        // [512][128] f16
#define WS_WIH1  163840u       // [512][128] f16
#define WS_WHH1  294912u       // [512][128] f16
#define WS_B0    425984u       // [512] f32 (b_ih0 + b_hh0)
#define WS_B1    428032u       // [512] f32 (b_ih1 + b_hh1)

// ---- CDNA5 async global->LDS copy (ASYNCcnt path) --------------------------
__device__ __forceinline__ void async_copy_b128(uint32_t lds_byte_addr, const void* gsrc) {
  asm volatile("global_load_async_to_lds_b128 %0, %1, off"
               :: "v"(lds_byte_addr), "v"((uint64_t)(uintptr_t)gsrc)
               : "memory");
}
__device__ __forceinline__ void wait_async0() {
  asm volatile("s_wait_asynccnt 0" ::: "memory");
}

// ---- WMMA helpers ----------------------------------------------------------
__device__ __forceinline__ v8f wmma_f16(v16h a, v16h b, v8f c) {
  // D = A(16x32,f16) x B(32x16,f16) + C(16x16,f32)
  return __builtin_amdgcn_wmma_f32_16x16x32_f16(false, a, false, b, (short)0, c, false, false);
}

__device__ __forceinline__ v8f splat8(float v) {
  v8f r;
#pragma unroll
  for (int j = 0; j < 8; ++j) r[j] = v;
  return r;
}

// A fragment (16x32 f16) from an f16 [64][128] row-major buffer.
// Lane l: M = l&15; K-base kg = (l>>4)*8; VGPR0-3 <- K=kg..kg+7, VGPR4-7 <- K=kg+16..kg+23.
__device__ __forceinline__ v16h loadA_h(const _Float16* hb, int ptile, int lane, int kk) {
  const _Float16* base = hb + (size_t)(ptile * 16 + (lane & 15)) * DIMH + kk * 32 + ((lane >> 4) * 8);
  v8h lo = *(const v8h*)base;         // K = kg .. kg+7
  v8h hi = *(const v8h*)(base + 16);  // K = kg+16 .. kg+23
  v16h r;
#pragma unroll
  for (int i = 0; i < 8; ++i) { r[i] = lo[i]; r[i + 8] = hi[i]; }
  return r;
}

// B fragment (32x16 f16) from f16 row-major W[G][rowstride]; column n = gate row.
// Lane l: N = gbase+(l&15); K-base kg = kk*32+(l>>4)*8; elems 0-7 <- K=kg..kg+7, elems 8-15 <- K=kg+16..kg+23.
__device__ __forceinline__ v16h loadB_w(const _Float16* W, int rowstride, int gbase, int lane, int kk) {
  const _Float16* base = W + (size_t)(gbase + (lane & 15)) * rowstride + kk * 32 + ((lane >> 4) * 8);
  v8h lo = *(const v8h*)base;
  v8h hi = *(const v8h*)(base + 16);
  v16h r;
#pragma unroll
  for (int i = 0; i < 8; ++i) { r[i] = lo[i]; r[i + 8] = hi[i]; }
  return r;
}

__device__ __forceinline__ float sigmoidf_(float x) {
  return 1.0f / (1.0f + __expf(-x));
}
__device__ __forceinline__ float tanhf_(float x) {
  float t = fminf(fmaxf(2.0f * x, -30.0f), 30.0f);
  float e = __expf(t);
  return (e - 1.0f) / (e + 1.0f);
}

// x_t A-fragment: 4x float4 (b128) loads, convert, K padded 28->32
__device__ __forceinline__ v16h loadA_x(const float* xr, int lane) {
  const int kg = (lane >> 4) * 8;
  v4f f0 = *(const v4f*)(xr + kg);        // K = kg .. kg+3
  v4f f1 = *(const v4f*)(xr + kg + 4);    // K = kg+4 .. kg+7
  v4f f2 = *(const v4f*)(xr + kg + 16);   // K = kg+16 .. kg+19
  v4f f3 = {0.0f, 0.0f, 0.0f, 0.0f};      // K = kg+20 .. kg+23 (28..31 -> pad)
  if (lane < 16) f3 = *(const v4f*)(xr + kg + 20);
  v16h ax;
#pragma unroll
  for (int j = 0; j < 4; ++j) {
    ax[j]      = (_Float16)f0[j];
    ax[j + 4]  = (_Float16)f1[j];
    ax[j + 8]  = (_Float16)f2[j];
    ax[j + 12] = (_Float16)f3[j];
  }
  return ax;
}

// LSTM cell elementwise for one 16x16 tile; biases folded here (gate chains
// seed from WMMA inline-zero C). Updates c in place, stores h (f16) to LDS.
__device__ __forceinline__ void lstm_cell(const v8f* acc4, const float* bq, v8f& cst,
                                          _Float16* hbuf, int ptile, int lane, int s) {
  const int ncol = lane & 15;
#pragma unroll
  for (int j = 0; j < 8; ++j) {
    float iv = sigmoidf_(acc4[0][j] + bq[0]);
    float fv = sigmoidf_(acc4[1][j] + bq[1]);
    float gv = tanhf_(acc4[2][j] + bq[2]);
    float ov = sigmoidf_(acc4[3][j] + bq[3]);
    float c  = fv * cst[j] + iv * gv;
    cst[j] = c;
    float h  = ov * tanhf_(c);
    hbuf[(size_t)(ptile * 16 + (lane >> 4) * 8 + j) * DIMH + s * 16 + ncol] = (_Float16)h;
  }
}

// ---------------------------------------------------------------------------
// Prep kernel: f32 weights -> f16 in d_ws (pad D 28->32), fuse biases.
// ---------------------------------------------------------------------------
__global__ void convert_weights_kernel(const float* __restrict__ Wih0, const float* __restrict__ Whh0,
                                       const float* __restrict__ bih0, const float* __restrict__ bhh0,
                                       const float* __restrict__ Wih1, const float* __restrict__ Whh1,
                                       const float* __restrict__ bih1, const float* __restrict__ bhh1,
                                       char* __restrict__ ws) {
  int tid = blockIdx.x * blockDim.x + threadIdx.x;  // 65536 threads
  _Float16* dWih0 = (_Float16*)(ws + WS_WIH0);
  _Float16* dWhh0 = (_Float16*)(ws + WS_WHH0);
  _Float16* dWih1 = (_Float16*)(ws + WS_WIH1);
  _Float16* dWhh1 = (_Float16*)(ws + WS_WHH1);
  float*    db0   = (float*)(ws + WS_B0);
  float*    db1   = (float*)(ws + WS_B1);

  if (tid < 16384) {  // [512][32] padded
    int g = tid >> 5, k = tid & 31;
    dWih0[tid] = (k < 28) ? (_Float16)Wih0[g * 28 + k] : (_Float16)0.0f;
  }
  if (tid < 65536) {  // [512][128] each
    dWhh0[tid] = (_Float16)Whh0[tid];
    dWih1[tid] = (_Float16)Wih1[tid];
    dWhh1[tid] = (_Float16)Whh1[tid];
  }
  if (tid < 512) {
    db0[tid] = bih0[tid] + bhh0[tid];
    db1[tid] = bih1[tid] + bhh1[tid];
  }
}

// ---------------------------------------------------------------------------
// Main fused kernel: 512 threads = 16 waves; 64 batch rows per workgroup.
// ---------------------------------------------------------------------------
__global__ __launch_bounds__(512, 1)
void lstm_fused_kernel(const float* __restrict__ x,       // [B][784] f32
                       const char* __restrict__ ws,       // f16 weights + biases
                       const float* __restrict__ Wfc,     // [10][128] f32
                       const float* __restrict__ bfc,     // [10] f32
                       float* __restrict__ out) {         // [B][10] f32
  extern __shared__ char smem[];
  _Float16* sWhh0 = (_Float16*)(smem + LDS_WHH0);
  _Float16* sWhh1 = (_Float16*)(smem + LDS_WHH1);
  _Float16* h0buf = (_Float16*)(smem + LDS_H0);
  _Float16* h2buf = (_Float16*)(smem + LDS_H2);

  const _Float16* gWih0 = (const _Float16*)(ws + WS_WIH0);  // [512][32]
  const _Float16* gWih1 = (const _Float16*)(ws + WS_WIH1);  // [512][128]
  const float*    bias0 = (const float*)(ws + WS_B0);
  const float*    bias1 = (const float*)(ws + WS_B1);

  const int tid  = threadIdx.x;
  const int lane = tid & 31;
  const int wid  = tid >> 5;
  const int s    = wid & 7;        // gate/h column slice [16s, 16s+16)
  const int pg   = wid >> 3;       // row-tile pair {0,1}
  const int pt0  = pg * 2;         // first 16-row tile
  const int pt1  = pg * 2 + 1;     // second 16-row tile
  const int ncol = lane & 15;

  // ---- stage W_hh0 / W_hh1 (f16, 256 KB) into LDS with async copies -------
#pragma unroll 4
  for (int r = 0; r < 16; ++r) {
    uint32_t off = (uint32_t)(r * 8192 + tid * 16);
    async_copy_b128((uint32_t)(uintptr_t)(smem + LDS_WHH0 + off), ws + WS_WHH0 + off);
    async_copy_b128((uint32_t)(uintptr_t)(smem + LDS_WHH1 + off), ws + WS_WHH1 + off);
  }
  // zero h0/h2 state buffers (32 KB = 2048 uint4) while copies are in flight
  {
    uint4 zz; zz.x = zz.y = zz.z = zz.w = 0u;
    ((uint4*)(smem + LDS_H0))[tid]        = zz;
    ((uint4*)(smem + LDS_H0))[tid + 512]  = zz;
    ((uint4*)(smem + LDS_H0))[tid + 1024] = zz;
    ((uint4*)(smem + LDS_H0))[tid + 1536] = zz;
  }
  wait_async0();
  __syncthreads();

  // fused biases for this wave's 4 gate columns (scalar per lane; applied in
  // the elementwise phase, NOT as WMMA C seeds)
  float bL0[4], bL1[4];
#pragma unroll
  for (int q = 0; q < 4; ++q) {
    bL0[q] = bias0[q * 128 + s * 16 + ncol];
    bL1[q] = bias1[q * 128 + s * 16 + ncol];
  }

  const v8f vzero = {};  // WMMA C = inline 0 (no register cost)

  v8f c0a = splat8(0.0f), c0b = splat8(0.0f);  // layer-0 cell state (2 row tiles)
  v8f c1a = splat8(0.0f), c1b = splat8(0.0f);  // layer-1 cell state

  const float* xrow0 = x + ((size_t)blockIdx.x * 64 + pt0 * 16 + ncol) * 784;
  const float* xrow1 = x + ((size_t)blockIdx.x * 64 + pt1 * 16 + ncol) * 784;

  for (int t = 0; t < T_STEPS; ++t) {
    // ================= layer 0: gates = x_t@Wih0^T + h0@Whh0^T (+b later) ==
    v8f acc0[4], acc1[4];
    {
      v16h ax0 = loadA_x(xrow0 + t * 28, lane);
      v16h ax1 = loadA_x(xrow1 + t * 28, lane);
#pragma unroll
      for (int q = 0; q < 4; ++q) {
        v16h b = loadB_w(gWih0, 32, q * 128 + s * 16, lane, 0);
        acc0[q] = wmma_f16(ax0, b, vzero);   // C = inline 0
        acc1[q] = wmma_f16(ax1, b, vzero);
      }
    }
#pragma unroll
    for (int kk = 0; kk < 4; ++kk) {
      v16h a0 = loadA_h(h0buf, pt0, lane, kk);
      v16h a1 = loadA_h(h0buf, pt1, lane, kk);
#pragma unroll
      for (int q = 0; q < 4; ++q) {
        v16h b = loadB_w(sWhh0, DIMH, q * 128 + s * 16, lane, kk);
        acc0[q] = wmma_f16(a0, b, acc0[q]);
        acc1[q] = wmma_f16(a1, b, acc1[q]);
      }
    }
    __syncthreads();  // all waves done reading h0_{t-1}

    lstm_cell(acc0, bL0, c0a, h0buf, pt0, lane, s);
    lstm_cell(acc1, bL0, c0b, h0buf, pt1, lane, s);
    __syncthreads();  // h0_t visible

    // ================= layer 1: gates = h0_t@Wih1^T + h2@Whh1^T (+b later) =
    {
      v16h a0 = loadA_h(h0buf, pt0, lane, 0);
      v16h a1 = loadA_h(h0buf, pt1, lane, 0);
#pragma unroll
      for (int q = 0; q < 4; ++q) {
        v16h b = loadB_w(gWih1, DIMH, q * 128 + s * 16, lane, 0);  // L2/WGP$-resident
        acc0[q] = wmma_f16(a0, b, vzero);
        acc1[q] = wmma_f16(a1, b, vzero);
      }
    }
#pragma unroll
    for (int kk = 1; kk < 4; ++kk) {
      v16h a0 = loadA_h(h0buf, pt0, lane, kk);
      v16h a1 = loadA_h(h0buf, pt1, lane, kk);
#pragma unroll
      for (int q = 0; q < 4; ++q) {
        v16h b = loadB_w(gWih1, DIMH, q * 128 + s * 16, lane, kk);
        acc0[q] = wmma_f16(a0, b, acc0[q]);
        acc1[q] = wmma_f16(a1, b, acc1[q]);
      }
    }
#pragma unroll
    for (int kk = 0; kk < 4; ++kk) {
      v16h a0 = loadA_h(h2buf, pt0, lane, kk);
      v16h a1 = loadA_h(h2buf, pt1, lane, kk);
#pragma unroll
      for (int q = 0; q < 4; ++q) {
        v16h b = loadB_w(sWhh1, DIMH, q * 128 + s * 16, lane, kk);
        acc0[q] = wmma_f16(a0, b, acc0[q]);
        acc1[q] = wmma_f16(a1, b, acc1[q]);
      }
    }
    __syncthreads();  // all waves done reading h2_{t-1}

    lstm_cell(acc0, bL1, c1a, h2buf, pt0, lane, s);
    lstm_cell(acc1, bL1, c1b, h2buf, pt1, lane, s);
    __syncthreads();  // h2_t visible
  }

  // ================= FC head: out = h2_{T-1} @ Wfc^T + bfc =================
  for (int o = tid; o < 640; o += 512) {  // 64 rows x 10 classes
    int r  = o / 10;
    int cc = o - r * 10;
    float a = bfc[cc];
    const float* wr = Wfc + cc * 128;
#pragma unroll 8
    for (int j = 0; j < 128; ++j) a += (float)h2buf[r * DIMH + j] * wr[j];
    out[((size_t)blockIdx.x * 64 + r) * 10 + cc] = a;
  }
}

// ---------------------------------------------------------------------------
extern "C" void kernel_launch(void* const* d_in, const int* in_sizes, int n_in,
                              void* d_out, int out_size, void* d_ws, size_t ws_size,
                              hipStream_t stream) {
  (void)in_sizes; (void)n_in; (void)out_size; (void)ws_size;
  const float* x    = (const float*)d_in[0];
  const float* Wih0 = (const float*)d_in[1];
  const float* Whh0 = (const float*)d_in[2];
  const float* bih0 = (const float*)d_in[3];
  const float* bhh0 = (const float*)d_in[4];
  const float* Wih1 = (const float*)d_in[5];
  const float* Whh1 = (const float*)d_in[6];
  const float* bih1 = (const float*)d_in[7];
  const float* bhh1 = (const float*)d_in[8];
  const float* Wfc  = (const float*)d_in[9];
  const float* bfc  = (const float*)d_in[10];
  float* out = (float*)d_out;
  char*  ws  = (char*)d_ws;

  convert_weights_kernel<<<256, 256, 0, stream>>>(Wih0, Whh0, bih0, bhh0,
                                                  Wih1, Whh1, bih1, bhh1, ws);

  (void)hipFuncSetAttribute((const void*)lstm_fused_kernel,
                            hipFuncAttributeMaxDynamicSharedMemorySize,
                            (int)LDS_TOTAL);
  lstm_fused_kernel<<<512, 512, LDS_TOTAL, stream>>>(x, ws, Wfc, bfc, out);
}
